// AGFAttention_9216999818002
// MI455X (gfx1250) — matrix operand order repaired and checked by reading, not compiled, not run on hardware
//
#include <hip/hip_runtime.h>
#include <hip/hip_bf16.h>
#include <stdint.h>

// ---------------------------------------------------------------------------
// AGF attention (top-64 masked softmax + monomial polynomial filter) for
// MI455X / gfx1250.  Matmuls: v_wmma_f32_16x16x32_f16 (f16 in, f32 accum).
// GEMM tiles staged with GLOBAL_LOAD_ASYNC_TO_LDS_B128 (ASYNCcnt DMA),
// double-buffered.  wave32 / WGP model throughout.
// ---------------------------------------------------------------------------

typedef _Float16 f16;
typedef __attribute__((ext_vector_type(16))) _Float16 v16h;
typedef __attribute__((ext_vector_type(8)))  float    v8f;

union Frag16 { v16h v; uint4 q[2]; };

// D = A(16x32) * B(32x16) + C, f32 accumulate
__device__ __forceinline__ v8f wmma_f16(v16h a, v16h b, v8f c) {
    return __builtin_amdgcn_wmma_f32_16x16x32_f16(
        /*neg_a=*/false, a, /*neg_b=*/false, b,
        /*c_mod=*/(short)0, c, /*reuse_a=*/false, /*reuse_b=*/false);
}

// CDNA5 async DMA: global -> LDS, 16 bytes per lane.  GV mode (SADDR=off).
__device__ __forceinline__ void async_copy_b128(unsigned lds_byte_off,
                                                const void* gptr) {
    asm volatile("global_load_async_to_lds_b128 %0, %1, off"
                 :: "v"(lds_byte_off),
                    "v"((unsigned long long)(size_t)gptr)
                 : "memory");
}
__device__ __forceinline__ void wait_async0() {
    asm volatile("s_wait_asynccnt 0x0" ::: "memory");
}

// A-fragment (16x32 f16), row-major storage, stride in halfs.
// lane<16 -> M=lane, K = {0..7, 16..23}; lane>=16 -> K = {8..15, 24..31}
__device__ __forceinline__ v16h load_a_frag(const f16* base, int stride, int ll, int hi) {
    Frag16 f;
    const f16* p = base + ll * stride;
    f.q[0] = *(const uint4*)(p + hi * 8);
    f.q[1] = *(const uint4*)(p + 16 + hi * 8);
    return f.v;
}

// B-fragment (32x16 f16) from K-major storage BT[n][k], stride in halfs.
// lane<16 -> N=lane, K = {0..7, 16..23}; lane>=16 -> N=lane-16, K = {8..15, 24..31}
__device__ __forceinline__ v16h load_bt_frag(const f16* bt, int stride, int ll, int hi) {
    Frag16 f;
    const f16* p = bt + ll * stride;
    f.q[0] = *(const uint4*)(p + hi * 8);
    f.q[1] = *(const uint4*)(p + 16 + hi * 8);
    return f.v;
}

// Order-preserving float<->uint key (top-k bisection)
__device__ __forceinline__ unsigned f2key(float s) {
    unsigned u = __float_as_uint(s);
    return (u & 0x80000000u) ? ~u : (u | 0x80000000u);
}
__device__ __forceinline__ float key2f(unsigned k) {
    unsigned u = (k & 0x80000000u) ? (k ^ 0x80000000u) : ~k;
    return __uint_as_float(u);
}

// ---------------------------------------------------------------------------
// float -> f16 conversion (4-wide) and 512-row transpose-convert
// ---------------------------------------------------------------------------
__global__ void cvt_f32_f16_kernel(const float* __restrict__ in,
                                   f16* __restrict__ out, int n4) {
    int i = blockIdx.x * blockDim.x + threadIdx.x;
    if (i < n4) {
        float4 f = ((const float4*)in)[i];
        f16* o = out + i * 4;
        o[0] = (f16)f.x; o[1] = (f16)f.y; o[2] = (f16)f.z; o[3] = (f16)f.w;
    }
}

// out[c*512 + r] = (f16) in[r*C + c]   (weights: (512 x C) -> K-major (C x 512))
__global__ void transpose_cvt_kernel(const float* __restrict__ in,
                                     f16* __restrict__ out, int C, int total) {
    int i = blockIdx.x * blockDim.x + threadIdx.x;
    if (i < total) {
        int c = i >> 9, r = i & 511;
        out[i] = (f16)in[r * C + c];
    }
}

// ---------------------------------------------------------------------------
// Double-buffered async WMMA GEMM mainloop.
// A: (rows x 512) f16 row-major.  Wt: (cols x 512) f16 K-major.
// Block = 256 threads (8 waves); tile 64x64; K-chunk 32; ping-pong LDS.
// ---------------------------------------------------------------------------
#define TS 2560   // halfs per LDS tile buffer (64 rows * 40 stride)

__device__ __forceinline__ void gemm_mainloop(
        const f16* __restrict__ A, const f16* __restrict__ Wt,
        int m0, int n0, int Kdim,
        f16* AsBuf, f16* BtBuf, unsigned asBase, unsigned btBase,
        int tid, int ll, int hi, int rg, int cg,
        v8f& acc0, v8f& acc1) {
    const int r = tid >> 2, p = tid & 3;                  // 64 x 4 copy map
    const f16* aSrc = A  + (size_t)(m0 + r) * Kdim + p * 8;
    const f16* bSrc = Wt + (size_t)(n0 + r) * Kdim + p * 8;
    const unsigned ldOff = (unsigned)((r * 40 + p * 8) * 2);
    const int nsteps = Kdim / 32;

    // prologue: stage chunk 0 into buffer 0
    async_copy_b128(asBase + ldOff, aSrc);
    async_copy_b128(btBase + ldOff, bSrc);

    for (int i = 0; i < nsteps; ++i) {
        wait_async0();          // our buffer (and any prefetch) landed
        __syncthreads();        // everyone's DMA landed
        if (i + 1 < nsteps) {   // prefetch next chunk into other buffer
            unsigned nb = ((i + 1) & 1) * (TS * 2);
            async_copy_b128(asBase + nb + ldOff, aSrc + (i + 1) * 32);
            async_copy_b128(btBase + nb + ldOff, bSrc + (i + 1) * 32);
        }
        const f16* as = AsBuf + (i & 1) * TS;
        const f16* bt = BtBuf + (i & 1) * TS;
        v16h a  = load_a_frag(as + (rg * 16) * 40, 40, ll, hi);
        v16h b0 = load_bt_frag(bt + (cg * 32) * 40, 40, ll, hi);
        v16h b1 = load_bt_frag(bt + (cg * 32 + 16) * 40, 40, ll, hi);
        acc0 = wmma_f16(a, b0, acc0);
        acc1 = wmma_f16(a, b1, acc1);
    }
}

// QKV GEMM: (8192x512) @ Wqkvt(1536x512 K-major) + bias, scatter to heads.
// q,k -> f16 [bh][n][64];  v -> f16 d-major [bh][64][2048]
__global__ __launch_bounds__(256)
void gemm_qkv_kernel(const f16* __restrict__ A, const f16* __restrict__ Wt,
                     const float* __restrict__ bias,
                     f16* __restrict__ qh, f16* __restrict__ kh,
                     f16* __restrict__ vt) {
    __shared__ __align__(16) f16 AsBuf[2 * TS];
    __shared__ __align__(16) f16 BtBuf[2 * TS];
    const int m0 = blockIdx.x * 64, n0 = blockIdx.y * 64;
    const int tid = threadIdx.x;
    const int wave = tid >> 5, lane = tid & 31, ll = lane & 15, hi = lane >> 4;
    const int rg = wave & 3, cg = wave >> 2;
    v8f acc0 = {}; v8f acc1 = {};

    gemm_mainloop(A, Wt, m0, n0, 512, AsBuf, BtBuf,
                  (unsigned)(size_t)(void*)AsBuf, (unsigned)(size_t)(void*)BtBuf,
                  tid, ll, hi, rg, cg, acc0, acc1);

    const int row_base = m0 + rg * 16;
#pragma unroll
    for (int t = 0; t < 2; ++t) {
        v8f c = t ? acc1 : acc0;
        int gc = n0 + cg * 32 + t * 16 + ll;          // 0..1535
        float bv = bias[gc];
        int which = gc >> 9, h = (gc >> 6) & 7, d = gc & 63;
#pragma unroll
        for (int r = 0; r < 8; ++r) {
            int row = row_base + r + hi * 8;          // 0..8191
            int bb = row >> 11, nn = row & 2047;
            int bh = bb * 8 + h;
            float val = c[r] + bv;
            if (which == 0)
                qh[(((size_t)bh * 2048) + nn) * 64 + d] = (f16)val;
            else if (which == 1)
                kh[(((size_t)bh * 2048) + nn) * 64 + d] = (f16)val;
            else
                vt[((size_t)bh * 64 + d) * 2048 + nn] = (f16)val;   // d-major
        }
    }
}

// Output GEMM: aout(8192x512) @ Woutt(512x512 K-major) + bout -> f32 out
__global__ __launch_bounds__(256)
void gemm_out_kernel(const f16* __restrict__ A, const f16* __restrict__ Wt,
                     const float* __restrict__ bias, float* __restrict__ out) {
    __shared__ __align__(16) f16 AsBuf[2 * TS];
    __shared__ __align__(16) f16 BtBuf[2 * TS];
    const int m0 = blockIdx.x * 64, n0 = blockIdx.y * 64;
    const int tid = threadIdx.x;
    const int wave = tid >> 5, lane = tid & 31, ll = lane & 15, hi = lane >> 4;
    const int rg = wave & 3, cg = wave >> 2;
    v8f acc0 = {}; v8f acc1 = {};

    gemm_mainloop(A, Wt, m0, n0, 512, AsBuf, BtBuf,
                  (unsigned)(size_t)(void*)AsBuf, (unsigned)(size_t)(void*)BtBuf,
                  tid, ll, hi, rg, cg, acc0, acc1);

    const int row_base = m0 + rg * 16;
#pragma unroll
    for (int t = 0; t < 2; ++t) {
        v8f c = t ? acc1 : acc0;
        int gc = n0 + cg * 32 + t * 16 + ll;
        float bv = bias[gc];
#pragma unroll
        for (int r = 0; r < 8; ++r) {
            int row = row_base + r + hi * 8;
            out[(size_t)row * 512 + gc] = c[r] + bv;
        }
    }
}

// ---------------------------------------------------------------------------
// Fused attention iteration (launched 3x):
//   scores = (Q Kh^T)*scale -> top-64 mask -> softmax -> P (f16, LDS)
//   vout^T = (P @ vin)^T ; res (+)= alpha_i * vout ; last: aout = res
// V buffers are d-major [bh][64][2048] f16 so the attn@V B-fragment is a
// contiguous global load (no LDS staging, no transpose).
// 1 block = 512 threads = 16 waves per (b,h,16-row tile).
// ---------------------------------------------------------------------------
#define SC_BYTES   (16 * 2048 * 4)        // 131072 (scores; aliased by partials)
#define PB_STRIDE  2056
#define PB_BYTES   (16 * PB_STRIDE * 2)   // 65792
#define ATTN_SMEM  (SC_BYTES + PB_BYTES)  // 196864 <= 320KB/WGP
#define PT_STRIDE  65                     // padded partials stride (floats)

__global__ __launch_bounds__(512)
void attn_iter_kernel(const f16* __restrict__ Q, const f16* __restrict__ K,
                      const f16* __restrict__ Vin, f16* __restrict__ Vout,
                      float* __restrict__ Res, f16* __restrict__ Aout,
                      const float* __restrict__ alphas_raw,
                      int iter, int is_first, int is_last) {
    extern __shared__ __align__(16) char smem[];
    float* sc       = (float*)smem;             // 16 x 2048 scores / e-values
    f16*   pb       = (f16*)(smem + SC_BYTES);  // 16 x PB_STRIDE probabilities
    float* partials = sc;                       // alias (sc dead in phase 3)

    const int gid  = blockIdx.x;
    const int tile = gid & 127;
    const int bh   = gid >> 7;                  // 0..31
    const int h    = bh & 7;
    const int row0 = tile * 16;

    const f16* Qh = Q   + (size_t)bh * 2048 * 64;
    const f16* Kh = K   + (size_t)bh * 2048 * 64;
    const f16* Vt = Vin + (size_t)bh * 64 * 2048;   // d-major

    const int tid = threadIdx.x;
    const int wave = tid >> 5, lane = tid & 31, ll = lane & 15, hi = lane >> 4;

    // -------- phase 1: scores = Q Kh^T * scale ------------------------------
    Frag16 A0, A1;
    {
        const f16* qrow = Qh + (size_t)(row0 + ll) * 64;
        A0.q[0] = *(const uint4*)(qrow + hi * 8);
        A0.q[1] = *(const uint4*)(qrow + 16 + hi * 8);
        A1.q[0] = *(const uint4*)(qrow + 32 + hi * 8);
        A1.q[1] = *(const uint4*)(qrow + 48 + hi * 8);
    }
    {
        const f16* kbase = Kh + (size_t)ll * 64;    // + t*1024 halfs per tile
        for (int t = wave; t < 128; t += 16) {
            const f16* krow = kbase + t * (16 * 64);
            if (t + 16 < 128)
                __builtin_prefetch(kbase + (t + 16) * (16 * 64), 0, 3);
            Frag16 B0, B1;
            B0.q[0] = *(const uint4*)(krow + hi * 8);
            B0.q[1] = *(const uint4*)(krow + 16 + hi * 8);
            B1.q[0] = *(const uint4*)(krow + 32 + hi * 8);
            B1.q[1] = *(const uint4*)(krow + 48 + hi * 8);
            v8f c = {};
            c = wmma_f16(A0.v, B0.v, c);
            c = wmma_f16(A1.v, B1.v, c);
#pragma unroll
            for (int r = 0; r < 8; ++r)
                sc[(r + hi * 8) * 2048 + t * 16 + ll] = c[r] * 0.125f; // DH^-0.5
        }
    }
    __syncthreads();

    // -------- phase 2: top-64 + softmax, one wave per row -------------------
    {
        float* srow = sc + wave * 2048;
        unsigned keys[64];
        unsigned kmax = 0u;
#pragma unroll
        for (int i = 0; i < 64; ++i) {
            unsigned kk = f2key(srow[lane + i * 32]);
            keys[i] = kk;
            kmax = (kk > kmax) ? kk : kmax;
        }
        for (int o = 16; o; o >>= 1) {
            unsigned other = (unsigned)__shfl_xor((int)kmax, o, 32);
            kmax = (other > kmax) ? other : kmax;
        }
        // bitwise bisection: largest T with count(key >= T) >= 64 == 64th key
        unsigned thr = 0u;
        for (int bit = 31; bit >= 0; --bit) {
            unsigned cand = thr | (1u << bit);
            int c = 0;
#pragma unroll
            for (int i = 0; i < 64; ++i) c += (keys[i] >= cand) ? 1 : 0;
            for (int o = 16; o; o >>= 1) c += __shfl_xor(c, o, 32);
            if (c >= 64) thr = cand;
        }
        const float maxv = key2f(kmax);
        float sum = 0.f;
#pragma unroll
        for (int i = 0; i < 64; ++i) {
            float e = 0.f;
            if (keys[i] >= thr) e = __expf(key2f(keys[i]) - maxv);
            srow[lane + i * 32] = e;     // masked -> exactly 0
            sum += e;
        }
        for (int o = 16; o; o >>= 1) sum += __shfl_xor(sum, o, 32);
        const float inv = 1.0f / sum;
        f16* prow = pb + wave * PB_STRIDE;
#pragma unroll
        for (int i = 0; i < 64; ++i)
            prow[lane + i * 32] = (f16)(srow[lane + i * 32] * inv);
    }
    __syncthreads();

    // -------- phase 3: out16x64 = P @ V, K=2048 split over 16 waves ---------
    v8f acc[4] = {{}, {}, {}, {}};
    {
        const f16* pbase = pb + ll * PB_STRIDE;     // A: probs, from LDS
        const f16* vbase = Vt + (size_t)ll * 2048 + hi * 8;  // B: V^T, global
        for (int cix = 0; cix < 4; ++cix) {
            const int k0 = (wave * 4 + cix) * 32;
            Frag16 a;
            a.q[0] = *(const uint4*)(pbase + k0 + hi * 8);
            a.q[1] = *(const uint4*)(pbase + k0 + 16 + hi * 8);
#pragma unroll
            for (int t = 0; t < 4; ++t) {
                const f16* vp = vbase + t * (16 * 2048) + k0;
                Frag16 b;
                b.q[0] = *(const uint4*)(vp);
                b.q[1] = *(const uint4*)(vp + 16);
                acc[t] = wmma_f16(a.v, b.v, acc[t]);
            }
        }
    }
#pragma unroll
    for (int t = 0; t < 4; ++t)
#pragma unroll
        for (int r = 0; r < 8; ++r)
            partials[(wave * 16 + r + hi * 8) * PT_STRIDE + t * 16 + ll] = acc[t][r];
    __syncthreads();

    // -------- reduce 16 partials; write vout^T / res / aout -----------------
    const float araw  = alphas_raw[iter * 8 + h];
    const float alpha = 0.5f * araw * (1.0f + erff(araw * 0.70710678118654752f));
    for (int o = tid; o < 1024; o += 512) {
        const int d = o >> 4, mr = o & 15;          // 16 rows contiguous/16 thr
        float s = 0.f;
#pragma unroll
        for (int w = 0; w < 16; ++w) s += partials[(w * 16 + mr) * PT_STRIDE + d];
        const int row = row0 + mr;
        const size_t didx = ((size_t)bh * 64 + d) * 2048 + row;  // d-major
        Vout[didx] = (f16)s;
        const float rv = (is_first ? 0.f : Res[didx]) + alpha * s;
        Res[didx] = rv;
        if (is_last) {
            const int b = bh >> 3;
            Aout[((size_t)b * 2048 + row) * 512 + h * 64 + d] = (f16)rv;
        }
    }
}

// ---------------------------------------------------------------------------
// host launcher
// ---------------------------------------------------------------------------
extern "C" void kernel_launch(void* const* d_in, const int* in_sizes, int n_in,
                              void* d_out, int out_size, void* d_ws, size_t ws_size,
                              hipStream_t stream) {
    (void)in_sizes; (void)n_in; (void)out_size; (void)ws_size;
    const float* x      = (const float*)d_in[0];   // (4,2048,512)
    const float* Wqkv   = (const float*)d_in[1];   // (512,1536)
    const float* bqkv   = (const float*)d_in[2];   // (1536,)
    const float* Wout   = (const float*)d_in[3];   // (512,512)
    const float* bout   = (const float*)d_in[4];   // (512,)
    const float* alphas = (const float*)d_in[5];   // (3,8)
    float* out = (float*)d_out;

    char* ws = (char*)d_ws;
    size_t off = 0;
    auto alloc = [&](size_t bytes) {
        void* p = ws + off;
        off = (off + bytes + 255) & ~(size_t)255;
        return p;
    };
    f16*   xh     = (f16*)alloc(4ull * 2048 * 512 * 2);   // x f16 row-major
    f16*   wqkvt  = (f16*)alloc(1536ull * 512 * 2);       // Wqkv^T (K-major)
    f16*   woutt  = (f16*)alloc(512ull * 512 * 2);        // Wout^T (K-major)
    f16*   qh     = (f16*)alloc(32ull * 2048 * 64 * 2);   // [bh][n][d]
    f16*   kh     = (f16*)alloc(32ull * 2048 * 64 * 2);   // [bh][n][d]
    f16*   vt0    = (f16*)alloc(32ull * 64 * 2048 * 2);   // v^T  [bh][d][n]
    f16*   vt1    = (f16*)alloc(32ull * 64 * 2048 * 2);
    f16*   vt2    = (f16*)alloc(32ull * 64 * 2048 * 2);
    float* res    = (float*)alloc(32ull * 64 * 2048 * 4); // d-major fp32
    f16*   aout   = (f16*)alloc(4ull * 2048 * 512 * 2);   // (b,n,inner) f16

    // conversions / weight pre-transposes (one-time, tiny vs. 23.3 TB/s)
    cvt_f32_f16_kernel<<<4096, 256, 0, stream>>>(x, xh, 1048576);
    transpose_cvt_kernel<<<3072, 256, 0, stream>>>(Wqkv, wqkvt, 1536, 786432);
    transpose_cvt_kernel<<<1024, 256, 0, stream>>>(Wout, woutt, 512, 262144);

    // qkv = x @ Wqkv + bqkv, scattered to head layouts (v stored d-major)
    gemm_qkv_kernel<<<dim3(128, 24), 256, 0, stream>>>(xh, wqkvt, bqkv, qh, kh, vt0);

    // polynomial filter: 3 fused attention iterations (global sync between)
    attn_iter_kernel<<<4096, 512, ATTN_SMEM, stream>>>(qh, kh, vt0, vt1, res, aout, alphas, 0, 1, 0);
    attn_iter_kernel<<<4096, 512, ATTN_SMEM, stream>>>(qh, kh, vt1, vt2, res, aout, alphas, 1, 0, 0);
    attn_iter_kernel<<<4096, 512, ATTN_SMEM, stream>>>(qh, kh, vt2, vt0, res, aout, alphas, 2, 0, 1);

    // out = aout @ Wout + bout
    gemm_out_kernel<<<dim3(128, 8), 256, 0, stream>>>(aout, woutt, bout, out);
}
